// MPNN_61426622268105
// MI455X (gfx1250) — matrix operand order repaired and checked by reading, not compiled, not run on hardware
//
#include <hip/hip_runtime.h>
#include <hip/hip_bf16.h>
#include <math.h>

// ---------------------------------------------------------------------------
// MPNN (REANN-style) for MI455X / gfx1250, wave32.
//   - edge kernels: E-parallel, hardware global_atomic_add_f32 into L2-resident
//     per-atom arrays; float4 gathers (16B-aligned padded rows)
//   - tiny radial/emb MLPs: scalar per-atom (8-wide, not worth matrix tiles)
//   - big [24->64->64->nout] MLPs: v_wmma_f32_16x16x32_f16; weights and
//     activations staged in LDS in *fragment order* so every WMMA operand is
//     two ds_load_b128 (no u16 gathers / b16 packing)
// ---------------------------------------------------------------------------

#define NWAVE   8
#define NORB    24      // (MAXL+1)*NWAVE
#define NCOEFF  17      // 2*NWAVE+1 logical outputs
#define CSTRIDE 24      // padded coeff row: [0..7]=dis, [8..15]=cart, [16]=center
#define NITER   3

typedef __attribute__((ext_vector_type(16))) _Float16 v16h;
typedef __attribute__((ext_vector_type(8)))  float    v8f;

__device__ __forceinline__ float silu_f(float x) { return x / (1.f + __expf(-x)); }

// hardware f32 atomic add (no return -> GLOBAL_ATOMIC_ADD_F32, STOREcnt)
__device__ __forceinline__ void atomic_add_f32(float* p, float v) {
    (void)__hip_atomic_fetch_add(p, v, __ATOMIC_RELAXED, __HIP_MEMORY_SCOPE_AGENT);
}

// logical coeff index (0..16) -> padded slot
__device__ __forceinline__ int coeff_slot(int o) {
    return (o < 8) ? o : ((o == 8) ? 16 : (o - 1));
}

// --------------------------- tiny scalar MLPs ------------------------------

__device__ __forceinline__ void ln_silu8(float* h, const float* g, const float* be) {
    float mu = 0.f;
    #pragma unroll
    for (int i = 0; i < 8; ++i) mu += h[i];
    mu *= 0.125f;
    float var = 0.f;
    #pragma unroll
    for (int i = 0; i < 8; ++i) { float d = h[i] - mu; var += d * d; }
    var *= 0.125f;
    float rs = rsqrtf(var + 1e-5f);
    #pragma unroll
    for (int i = 0; i < 8; ++i) h[i] = silu_f((h[i] - mu) * rs * g[i] + be[i]);
}

// coeff = nnmod(emb, species): [1 -> 8 -> 8] + head 8->17
__global__ void emb_kernel(const float* __restrict__ species, int N,
                           const float* W0, const float* b0,
                           const float* W1, const float* b1,
                           const float* g0, const float* be0,
                           const float* g1, const float* be1,
                           const float* Wo, const float* bo,
                           float* __restrict__ coeff) {
    int n = blockIdx.x * blockDim.x + threadIdx.x;
    if (n >= N) return;
    float x = species[n];
    float h[8];
    #pragma unroll
    for (int o = 0; o < 8; ++o) h[o] = x * W0[o] + b0[o];
    ln_silu8(h, g0, be0);
    float h2[8];
    #pragma unroll
    for (int o = 0; o < 8; ++o) {
        float s = b1[o];
        #pragma unroll
        for (int i = 0; i < 8; ++i) s += h[i] * W1[i * 8 + o];
        h2[o] = s;
    }
    ln_silu8(h2, g1, be1);
    float* cr = coeff + (size_t)n * CSTRIDE;
    #pragma unroll
    for (int o = 0; o < NCOEFF; ++o) {
        float s = bo[o];
        #pragma unroll
        for (int i = 0; i < 8; ++i) s += h2[i] * Wo[i * NCOEFF + o];
        cr[coeff_slot(o)] = s;
    }
}

// --------------------------- edge kernels ----------------------------------

__global__ void edge_setup_kernel(const float* __restrict__ cart,
                                  const int* __restrict__ nl,
                                  const float* __restrict__ shifts,
                                  const float* __restrict__ neigh_factor,
                                  int E,
                                  float* __restrict__ distvec,
                                  float* __restrict__ cutb) {
    int e = blockIdx.x * blockDim.x + threadIdx.x;
    if (e >= E) return;
    int c  = nl[e];
    int nb = nl[E + e];
    float dx = cart[nb * 3 + 0] - cart[c * 3 + 0] + shifts[e * 3 + 0];
    float dy = cart[nb * 3 + 1] - cart[c * 3 + 1] + shifts[e * 3 + 1];
    float dz = cart[nb * 3 + 2] - cart[c * 3 + 2] + shifts[e * 3 + 2];
    float dist = sqrtf(dx * dx + dy * dy + dz * dz);
    float t = 0.5f * __cosf(dist * 0.78539816339744831f) + 0.5f;  // pi/CUTOFF
    cutb[e] = t * t * neigh_factor[e];
    distvec[e * 3 + 0] = dx;
    distvec[e * 3 + 1] = dy;
    distvec[e * 3 + 2] = dz;
}

// segment_sum scatter: MP_dis (N x 8) and MP_cart_new (N x 3 x 8)
__global__ void edge_scatter_kernel(const int* __restrict__ nl,
                                    const float* __restrict__ distvec,
                                    const float* __restrict__ cutb,
                                    const float* __restrict__ coeff,
                                    const float* __restrict__ mpc_old,
                                    float* __restrict__ MP_dis,
                                    float* __restrict__ mpc_new,
                                    int E) {
    int e = blockIdx.x * blockDim.x + threadIdx.x;
    if (e >= E) return;
    int c  = nl[e];
    int nb = nl[E + e];
    float cut = cutb[e];

    const float4* ncv = (const float4*)(coeff + (size_t)nb * CSTRIDE);
    float4 w0 = ncv[0], w1 = ncv[1];   // dis coefficients (slots 0..7)
    float4 p0 = ncv[2], p1 = ncv[3];   // cart coefficients (slots 8..15)
    const float4* mov = (const float4*)(mpc_old + (size_t)nb * 24);
    float4 m0 = mov[0], m1 = mov[1], m2 = mov[2], m3 = mov[3], m4 = mov[4], m5 = mov[5];

    float wd[8] = {w0.x, w0.y, w0.z, w0.w, w1.x, w1.y, w1.z, w1.w};
    float pk[8] = {p0.x, p0.y, p0.z, p0.w, p1.x, p1.y, p1.z, p1.w};
    float mo[24] = {m0.x, m0.y, m0.z, m0.w, m1.x, m1.y, m1.z, m1.w,
                    m2.x, m2.y, m2.z, m2.w, m3.x, m3.y, m3.z, m3.w,
                    m4.x, m4.y, m4.z, m4.w, m5.x, m5.y, m5.z, m5.w};

    float* dis = MP_dis + (size_t)c * NWAVE;
    #pragma unroll
    for (int w = 0; w < NWAVE; ++w) atomic_add_f32(&dis[w], cut * wd[w]);

    float dv0 = distvec[e * 3 + 0], dv1 = distvec[e * 3 + 1], dv2 = distvec[e * 3 + 2];
    float* mn = mpc_new + (size_t)c * 24;
    #pragma unroll
    for (int w = 0; w < NWAVE; ++w) {
        atomic_add_f32(&mn[0 * 8 + w], cut * (dv0 * pk[w] + mo[0 * 8 + w]));
        atomic_add_f32(&mn[1 * 8 + w], cut * (dv1 * pk[w] + mo[1 * 8 + w]));
        atomic_add_f32(&mn[2 * 8 + w], cut * (dv2 * pk[w] + mo[2 * 8 + w]));
    }
}

// per-atom: radial MLP [8->8->8]+head, sph contraction, density accumulate
__global__ void atom_kernel(const float* __restrict__ MP_dis,
                            const float* __restrict__ coeff,
                            const float* __restrict__ mpc,
                            const float* Wr0, const float* br0,
                            const float* Wr1, const float* br1,
                            const float* gr0, const float* ber0,
                            const float* gr1, const float* ber1,
                            const float* Wro, const float* bro,
                            float* __restrict__ density, int N) {
    int n = blockIdx.x * blockDim.x + threadIdx.x;
    if (n >= N) return;
    float cen = coeff[(size_t)n * CSTRIDE + 16];
    float x[8];
    #pragma unroll
    for (int i = 0; i < 8; ++i) x[i] = MP_dis[(size_t)n * 8 + i] * cen;
    float h[8];
    #pragma unroll
    for (int o = 0; o < 8; ++o) {
        float s = br0[o];
        #pragma unroll
        for (int i = 0; i < 8; ++i) s += x[i] * Wr0[i * 8 + o];
        h[o] = s;
    }
    ln_silu8(h, gr0, ber0);
    float h2[8];
    #pragma unroll
    for (int o = 0; o < 8; ++o) {
        float s = br1[o];
        #pragma unroll
        for (int i = 0; i < 8; ++i) s += h[i] * Wr1[i * 8 + o];
        h2[o] = s;
    }
    ln_silu8(h2, gr1, ber1);
    float rad[8];
    #pragma unroll
    for (int o = 0; o < 8; ++o) {
        float s = bro[o];
        #pragma unroll
        for (int i = 0; i < 8; ++i) s += h2[i] * Wro[i * 8 + o];
        rad[o] = s;
    }
    const float* mp = mpc + (size_t)n * 24;
    float* dn = density + (size_t)n * NORB;
    #pragma unroll
    for (int w = 0; w < NWAVE; ++w) {
        float X = mp[0 * 8 + w], Y = mp[1 * 8 + w], Z = mp[2 * 8 + w];
        float r2 = X * X + Y * Y + Z * Z;
        float s0 = 0.28209479f;
        float s1 = 0.48860251f * Y, s2 = 0.48860251f * Z, s3 = 0.48860251f * X;
        float s4 = 1.09254843f * X * Y, s5 = 1.09254843f * Y * Z;
        float s6 = 0.31539157f * (3.f * Z * Z - r2);
        float s7 = 1.09254843f * X * Z;
        float s8 = 0.54627422f * (X * X - Y * Y);
        float a0 = s0 * s0;
        float a1 = s1 * s1 + s2 * s2 + s3 * s3;
        float a2 = s4 * s4 + s5 * s5 + s6 * s6 + s7 * s7 + s8 * s8;
        dn[0 * 8 + w] += rad[w] * a0;
        dn[1 * 8 + w] += rad[w] * a1;
        dn[2 * 8 + w] += rad[w] * a2;
    }
}

// --------------------------- WMMA big MLP ----------------------------------
//
// Fragment-order LDS: a 32x16 (KxN) B tile, or a 16x32 (MxK) A tile, is one
// v16h per lane (32B) -> operand load = 2x ds_load_b128, zero packing.
//   B tile element (lane, j): k = kbase + ((lane<16)?0:8) + ((j<8)?j:j+8), n = n0 + (lane&15)
//   A tile element (lane, j): M = lane&15, kb = (lane<16)?0:8,
//                             k = (j<8) ? kb+j : 16+kb+(j-8)

__device__ __forceinline__ v8f wmma16(v16h a, v16h b, v8f c) {
    return __builtin_amdgcn_wmma_f32_16x16x32_f16(false, a, false, b,
                                                  (short)0, c, false, false);
}

#define MLP_WAVES 4
#define MLP_BLK   (MLP_WAVES * 32)

__global__ __launch_bounds__(MLP_BLK)
void mlp_wmma_kernel(const float* __restrict__ X, int N,
                     const float* W1, const float* b1v,
                     const float* W2, const float* b2v,
                     const float* g1v, const float* be1v,
                     const float* g2v, const float* be2v,
                     const float* Wo, const float* bov, int nout,
                     float* __restrict__ coeff_out, int finalMode,
                     const float* __restrict__ center_factor,
                     float* __restrict__ out_scalar) {
    __shared__ v16h W1f[4][32];            // [nt]          (kchunk 0, rows 24..31 = 0)
    __shared__ v16h W2f[8][32];            // [kc*4 + nt]
    __shared__ v16h W3f[4][32];            // [kc*2 + nt]   (cols >= nout = 0)
    __shared__ float bias1[64], gam1[64], bet1[64];
    __shared__ float bias2[64], gam2[64], bet2[64];
    __shared__ float bias3[32];
    __shared__ float act32[MLP_WAVES][16][64];
    __shared__ v16h  afrag[MLP_WAVES][2][32];   // A fragments (K=64 -> 2 chunks)

    const int tid = threadIdx.x;

    // ---- stage weights straight into fragment order (once per block) ----
    for (int i = tid; i < 4 * 512; i += MLP_BLK) {          // W1: 24x64 pad K->32
        int t = i >> 9, r = i & 511, lane = r >> 4, j = r & 15;
        int k = ((lane < 16) ? 0 : 8) + ((j < 8) ? j : j + 8);
        int n = (t << 4) + (lane & 15);
        float v = (k < NORB) ? W1[k * 64 + n] : 0.f;
        ((_Float16*)&W1f[t][lane])[j] = (_Float16)v;
    }
    for (int i = tid; i < 8 * 512; i += MLP_BLK) {          // W2: 64x64
        int t = i >> 9, r = i & 511, lane = r >> 4, j = r & 15;
        int kc = t >> 2, nt = t & 3;
        int k = kc * 32 + ((lane < 16) ? 0 : 8) + ((j < 8) ? j : j + 8);
        int n = (nt << 4) + (lane & 15);
        ((_Float16*)&W2f[t][lane])[j] = (_Float16)W2[k * 64 + n];
    }
    for (int i = tid; i < 4 * 512; i += MLP_BLK) {          // W3: 64 x nout pad N->32
        int t = i >> 9, r = i & 511, lane = r >> 4, j = r & 15;
        int kc = t >> 1, nt = t & 1;
        int k = kc * 32 + ((lane < 16) ? 0 : 8) + ((j < 8) ? j : j + 8);
        int col = (nt << 4) + (lane & 15);
        float v = (col < nout) ? Wo[k * nout + col] : 0.f;
        ((_Float16*)&W3f[t][lane])[j] = (_Float16)v;
    }
    for (int i = tid; i < 64; i += MLP_BLK) {
        bias1[i] = b1v[i]; gam1[i] = g1v[i]; bet1[i] = be1v[i];
        bias2[i] = b2v[i]; gam2[i] = g2v[i]; bet2[i] = be2v[i];
    }
    for (int i = tid; i < 32; i += MLP_BLK) bias3[i] = (i < nout) ? bov[i] : 0.f;
    __syncthreads();

    const int wave = tid >> 5, lane = tid & 31;
    const int row0 = blockIdx.x * (MLP_WAVES * 16) + wave * 16;
    if (row0 >= N) return;              // wave-uniform: EXEC all-ones at every WMMA

    const int M   = lane & 15;
    const int kbA = (lane < 16) ? 0 : 8;
    const int hi  = (lane >= 16) ? 8 : 0;

    // ---- layer 1: X(16x24 pad 32) @ W1(32x64) ----
    v16h a1;
    {
        int row = row0 + M;
        bool ok = row < N;
        const float* xr = X + (size_t)row * NORB;
        #pragma unroll
        for (int j = 0; j < 16; ++j) {
            int k = (j < 8) ? (kbA + j) : (16 + kbA + j - 8);
            float v = (ok && k < NORB) ? xr[k] : 0.f;
            a1[j] = (_Float16)v;
        }
    }
    #pragma unroll
    for (int nt = 0; nt < 4; ++nt) {
        v8f c = {};
        c = wmma16(a1, W1f[nt][lane], c);
        #pragma unroll
        for (int r = 0; r < 8; ++r) act32[wave][r + hi][nt * 16 + M] = c[r];
    }

    // layernorm + silu, emitted directly as A fragments (lanes 0..15 own rows)
    #pragma unroll 1
    for (int layer = 0; layer < 2; ++layer) {
        const float* bias = layer ? bias2 : bias1;
        const float* g    = layer ? gam2  : gam1;
        const float* be   = layer ? bet2  : bet1;
        if (lane < 16) {
            float mu = 0.f;
            #pragma unroll
            for (int c = 0; c < 64; ++c) { float h = act32[wave][lane][c] + bias[c]; act32[wave][lane][c] = h; mu += h; }
            mu *= (1.f / 64.f);
            float var = 0.f;
            #pragma unroll
            for (int c = 0; c < 64; ++c) { float d = act32[wave][lane][c] - mu; var += d * d; }
            var *= (1.f / 64.f);
            float rs = rsqrtf(var + 1e-5f);
            #pragma unroll
            for (int c = 0; c < 64; ++c) {
                float hn = (act32[wave][lane][c] - mu) * rs * g[c] + be[c];
                _Float16 hv = (_Float16)silu_f(hn);
                int ch = c >> 5;          // kchunk
                int kk = c & 31;
                int hb = kk >> 3;         // 0..3
                int lane2 = lane + ((hb & 1) ? 16 : 0);
                int j = (hb == 0) ? kk : (hb == 1) ? kk - 8 : (hb == 2) ? kk - 8 : kk - 16;
                ((_Float16*)&afrag[wave][ch][lane2])[j] = hv;
            }
        }
        if (layer == 0) {
            // ---- layer 2: H(16x64) @ W2(64x64) ----
            v16h a20 = afrag[wave][0][lane];
            v16h a21 = afrag[wave][1][lane];
            #pragma unroll
            for (int nt = 0; nt < 4; ++nt) {
                v8f c = {};
                c = wmma16(a20, W2f[0 * 4 + nt][lane], c);
                c = wmma16(a21, W2f[1 * 4 + nt][lane], c);
                #pragma unroll
                for (int r = 0; r < 8; ++r) act32[wave][r + hi][nt * 16 + M] = c[r];
            }
        }
    }

    // ---- head: H(16x64) @ W3(64x32) ----
    {
        v16h a30 = afrag[wave][0][lane];
        v16h a31 = afrag[wave][1][lane];
        int ntiles = (nout + 15) / 16;
        #pragma unroll
        for (int nt = 0; nt < 2; ++nt) {
            if (nt >= ntiles) break;
            v8f c = {};
            c = wmma16(a30, W3f[0 * 2 + nt][lane], c);
            c = wmma16(a31, W3f[1 * 2 + nt][lane], c);
            int col = nt * 16 + M;
            if (!finalMode) {
                if (col < nout) {
                    int slot = coeff_slot(col);
                    #pragma unroll
                    for (int r = 0; r < 8; ++r) {
                        int row = row0 + r + hi;
                        if (row < N) coeff_out[(size_t)row * CSTRIDE + slot] = c[r] + bias3[col];
                    }
                }
            } else if (col == 0) {        // lanes 0 / 16 cover M = 0..7 / 8..15
                float acc = 0.f;
                #pragma unroll
                for (int r = 0; r < 8; ++r) {
                    int row = row0 + r + hi;
                    if (row < N) acc += (c[r] + bias3[0]) * center_factor[row];
                }
                atomic_add_f32(out_scalar, acc);
            }
        }
    }
}

// --------------------------- host orchestration ----------------------------
//
// Input layout (setup_inputs dict order; params pytree-flattened, inner dict
// keys sorted: emb, iter[0..2], out, radial; each nnmod flattens as
// hidden W0,b0,W1,b1 | ln g0,be0,g1,be1 | out Wo,bo):
//   0 cart 1 neighlist 2 shifts 3 center_factor 4 neigh_factor 5 species
//   6..15  emb      16..25 iter0   26..35 iter1   36..45 iter2
//   46..55 out head 56..65 radial

extern "C" void kernel_launch(void* const* d_in, const int* in_sizes, int n_in,
                              void* d_out, int out_size, void* d_ws, size_t ws_size,
                              hipStream_t stream) {
    (void)n_in; (void)out_size; (void)ws_size;
#define PF(i) ((const float*)d_in[(i)])
    const float* cart          = PF(0);
    const int*   nl            = (const int*)d_in[1];
    const float* shifts        = PF(2);
    const float* center_factor = PF(3);
    const float* neigh_factor  = PF(4);
    const float* species       = PF(5);
    const int N = in_sizes[0] / 3;
    const int E = in_sizes[1] / 2;

    // workspace carve (f32 elements, 16-element aligned so float4 casts hold)
    float* ws = (float*)d_ws;
    size_t off = 0;
    #define CARVE(name, nelem) float* name = ws + off; off += (((size_t)(nelem) + 15) & ~(size_t)15)
    CARVE(distvec, (size_t)E * 3);
    CARVE(cutb,    (size_t)E);
    CARVE(coeff,   (size_t)N * CSTRIDE);
    CARVE(MP_dis,  (size_t)N * 8);
    CARVE(mpcA,    (size_t)N * 24);
    CARVE(mpcB,    (size_t)N * 24);
    CARVE(density, (size_t)N * 24);
    #undef CARVE

    hipMemsetAsync(d_out,   0, sizeof(float), stream);
    hipMemsetAsync(density, 0, (size_t)N * 24 * sizeof(float), stream);
    hipMemsetAsync(mpcA,    0, (size_t)N * 24 * sizeof(float), stream);

    edge_setup_kernel<<<(E + 255) / 256, 256, 0, stream>>>(
        cart, nl, shifts, neigh_factor, E, distvec, cutb);

    emb_kernel<<<(N + 255) / 256, 256, 0, stream>>>(
        species, N, PF(6), PF(7), PF(8), PF(9), PF(10), PF(11), PF(12), PF(13),
        PF(14), PF(15), coeff);

    float* mpc_cur = mpcA;
    float* mpc_new = mpcB;
    for (int it = 0; it <= NITER; ++it) {
        hipMemsetAsync(MP_dis,  0, (size_t)N * 8  * sizeof(float), stream);
        hipMemsetAsync(mpc_new, 0, (size_t)N * 24 * sizeof(float), stream);

        edge_scatter_kernel<<<(E + 255) / 256, 256, 0, stream>>>(
            nl, distvec, cutb, coeff, mpc_cur, MP_dis, mpc_new, E);

        atom_kernel<<<(N + 255) / 256, 256, 0, stream>>>(
            MP_dis, coeff, mpc_new,
            PF(56), PF(57), PF(58), PF(59), PF(60), PF(61), PF(62), PF(63),
            PF(64), PF(65), density, N);

        const int base = (it < NITER) ? (16 + it * 10) : 46;
        const int nout = (it < NITER) ? NCOEFF : 1;
        const int rowsPerBlock = MLP_WAVES * 16;
        mlp_wmma_kernel<<<(N + rowsPerBlock - 1) / rowsPerBlock, MLP_BLK, 0, stream>>>(
            density, N,
            PF(base + 0), PF(base + 1), PF(base + 2), PF(base + 3),
            PF(base + 4), PF(base + 5), PF(base + 6), PF(base + 7),
            PF(base + 8), PF(base + 9), nout,
            coeff, (it == NITER) ? 1 : 0, center_factor, (float*)d_out);

        float* t = mpc_cur; mpc_cur = mpc_new; mpc_new = t;
    }
#undef PF
}
